// SwitchMoE_20323785244715
// MI455X (gfx1250) — matrix operand order repaired and checked by reading, not compiled, not run on hardware
//
#include <hip/hip_runtime.h>
#include <hip/hip_bf16.h>
#include <math.h>
#include <type_traits>

// ---------------------------------------------------------------------------
// Switch-MoE for MI455X (gfx1250): routed top-2 experts, bf16 WMMA FFN.
// N=4096 tokens, DIM=1024, E=8 experts, INNER=4096, OUT=1024, TOP_K=2.
// ---------------------------------------------------------------------------

#define NTOK  4096
#define DIM   1024
#define OUTD  1024
#define NEXP  8
#define INNER 4096
#define CAPACITY 4096.0f
#define EPSV  1e-6f

#define TM      64     // token tile per workgroup (~199KB of 320KB LDS)
#define CI      512    // INNER chunk held in LDS per pass
#define THREADS 512    // 16 waves -> 4 waves/SIMD -> ~256 VGPR budget
#define XPAD 8
#define HPAD 8

typedef __bf16 v16bf __attribute__((ext_vector_type(16)));
typedef float  v8f   __attribute__((ext_vector_type(8)));

// Branch-free exact-erf GELU (A&S 7.1.26, |err| < 1.5e-7).
__device__ __forceinline__ float gelu_erf(float v) {
    float xa = fabsf(v) * 0.7071067811865475f;
    float t  = __builtin_amdgcn_rcpf(1.0f + 0.3275911f * xa);
    float poly = t * (0.254829592f +
                 t * (-0.284496736f +
                 t * (1.421413741f +
                 t * (-1.453152027f +
                 t * 1.061405429f))));
    float er = copysignf(1.0f - poly * __expf(-xa * xa), v);
    return 0.5f * v * (1.0f + er);
}

// A-fragment (16x32 bf16) gather from LDS (vectorizes to ds_load_b128).
// M=lane%16, K = 16*half + j (contiguous, matches B fragments).
__device__ __forceinline__ v16bf lds_frag_a(const __bf16* base, int k0, int halfw) {
    v16bf a;
#pragma unroll
    for (int j = 0; j < 16; ++j) {
        int ka = (halfw << 4) + j;
        a[j] = base[k0 + ka];
    }
    return a;
}

// B-fragment (32x16) scalar gather from row-major f32 weights (fallback).
template <typename WT>
__device__ __forceinline__ v16bf glob_frag_b(const WT* col_ptr, int krow,
                                             int ld, int halfw) {
    v16bf b;
#pragma unroll
    for (int j = 0; j < 16; ++j) {
        int kb = (halfw << 4) + j;
        b[j] = (__bf16)col_ptr[(size_t)(krow + kb) * ld];
    }
    return b;
}

// ---------------------------------------------------------------------------
// K0: zero expert counters and the two per-slot contribution buffers.
// ---------------------------------------------------------------------------
__global__ void init_kernel(int* counts, float* buf, long long buf_elems) {
    long long i = (long long)blockIdx.x * blockDim.x + threadIdx.x;
    if (i < NEXP) counts[i] = 0;
    for (long long j = i; j < buf_elems; j += (long long)gridDim.x * blockDim.x)
        buf[j] = 0.0f;
}

// ---------------------------------------------------------------------------
// K0a: fp32 -> bf16 flat conversion (activations), float4 loads.
// ---------------------------------------------------------------------------
__global__ void cvt_bf16_kernel(const float* __restrict__ src,
                                __bf16* __restrict__ dst, long long n) {
    long long i = ((long long)blockIdx.x * blockDim.x + threadIdx.x) * 4;
    long long stride = (long long)gridDim.x * blockDim.x * 4;
    for (; i + 3 < n; i += stride) {
        float4 v = *(const float4*)(src + i);
        dst[i + 0] = (__bf16)v.x;
        dst[i + 1] = (__bf16)v.y;
        dst[i + 2] = (__bf16)v.z;
        dst[i + 3] = (__bf16)v.w;
    }
}

// ---------------------------------------------------------------------------
// K0b: fp32 -> bf16 weight conversion with WMMA-B tile swizzle.
// src: [E][K][C] f32 row-major.
// dst: [E][K/32][C/16] tiles of 512 bf16 in lane-major order:
//      tile[lane][j] = src[kt*32 + 16*(lane/16) + j][ct*16 + lane%16]
// so a wave's B fragment is one contiguous 32B load per lane (2x b128).
// One block (256 thr) per (e, kt) 32-row K-strip; C chunked by 1024 cols.
// ---------------------------------------------------------------------------
__global__ void cvt_tile_kernel(const float* __restrict__ src,
                                __bf16* __restrict__ dst, int K, int C) {
    __shared__ __bf16 tb[32][1024 + 8];    // 64.5 KB
    int kt = blockIdx.x % (K >> 5);
    int e  = blockIdx.x / (K >> 5);
    const float* s = src + ((size_t)e * K + (kt << 5)) * C;
    size_t strip_base = (((size_t)e * (K >> 5) + kt) * (C >> 4)) << 9;

    for (int c0 = 0; c0 < C; c0 += 1024) {
        // coalesced load+convert of 32 x 1024 f32 into LDS
        for (int i = threadIdx.x; i < 32 * 1024; i += 256) {
            int row = i >> 10, col = i & 1023;
            tb[row][col] = (__bf16)s[(size_t)row * C + c0 + col];
        }
        __syncthreads();
        // coalesced tile-order writes (consecutive threads -> consecutive dst)
        for (int i = threadIdx.x; i < 32 * 1024; i += 256) {
            int ctl  = i >> 9;                 // tile within chunk (0..63)
            int lane = (i >> 4) & 31;
            int j    = i & 15;
            int row  = ((lane >> 4) << 4) + j; // K within tile
            int col  = (ctl << 4) + (lane & 15);
            size_t ct = (size_t)(c0 >> 4) + ctl;
            dst[strip_base + (ct << 9) + ((size_t)lane << 4) + j] = tb[row][col];
        }
        __syncthreads();
    }
}

// ---------------------------------------------------------------------------
// K1: gate logits -> softmax -> top-2 mask.  One wave per token.
// ---------------------------------------------------------------------------
__global__ void gate_kernel(const float* __restrict__ gi,
                            const float* __restrict__ Wg,
                            const float* __restrict__ bg,
                            float* __restrict__ gmask) {
    int wave = threadIdx.x >> 5;
    int lane = threadIdx.x & 31;
    int n = blockIdx.x * 8 + wave;
    if (n >= NTOK) return;

    float acc[NEXP];
#pragma unroll
    for (int e = 0; e < NEXP; ++e) acc[e] = 0.0f;

    const float* row = gi + (size_t)n * DIM;
    for (int d = lane; d < DIM; d += 32) {
        float xv = row[d];
#pragma unroll
        for (int e = 0; e < NEXP; ++e) acc[e] += xv * Wg[d * NEXP + e];
    }
#pragma unroll
    for (int e = 0; e < NEXP; ++e)
        for (int off = 16; off > 0; off >>= 1)
            acc[e] += __shfl_xor(acc[e], off, 32);

    float z[NEXP], mx = -1e30f;
#pragma unroll
    for (int e = 0; e < NEXP; ++e) { z[e] = acc[e] + bg[e]; mx = fmaxf(mx, z[e]); }
    float s = 0.0f;
#pragma unroll
    for (int e = 0; e < NEXP; ++e) { z[e] = __expf(z[e] - mx); s += z[e]; }
    float inv = 1.0f / s;
#pragma unroll
    for (int e = 0; e < NEXP; ++e) z[e] *= inv;

    int i1 = 0;
#pragma unroll
    for (int e = 1; e < NEXP; ++e) if (z[e] > z[i1]) i1 = e;
    int i2 = (i1 == 0) ? 1 : 0;
#pragma unroll
    for (int e = 0; e < NEXP; ++e)
        if (e != i1 && e != i2 && z[e] > z[i2]) i2 = e;

    if (lane < NEXP)
        gmask[(size_t)n * NEXP + lane] =
            (lane == i1 || lane == i2) ? z[lane] : 0.0f;
}

// ---------------------------------------------------------------------------
// K2: per-expert load (denominator).  One block per expert.
// ---------------------------------------------------------------------------
__global__ void denom_kernel(const float* __restrict__ gmask,
                             float* __restrict__ denom) {
    __shared__ float red[256];
    int e = blockIdx.x;
    float s = 0.0f;
    for (int n = threadIdx.x; n < NTOK; n += 256) s += gmask[(size_t)n * NEXP + e];
    red[threadIdx.x] = s;
    __syncthreads();
    for (int o = 128; o > 0; o >>= 1) {
        if (threadIdx.x < o) red[threadIdx.x] += red[threadIdx.x + o];
        __syncthreads();
    }
    if (threadIdx.x == 0) denom[e] = red[0] + EPSV;
}

// ---------------------------------------------------------------------------
// K3: routing.  List order is atomic-nondeterministic, but GEMM rows are
// numerically independent, so the final output stays deterministic.
// ---------------------------------------------------------------------------
__global__ void route_kernel(const float* __restrict__ gmask,
                             const float* __restrict__ denom,
                             int* counts, int* list, float* wlist, int* slotArr) {
    int n = blockIdx.x * 256 + threadIdx.x;
    if (n >= NTOK) return;
    int slot = 0;
#pragma unroll
    for (int e = 0; e < NEXP; ++e) {
        float g = gmask[(size_t)n * NEXP + e];
        if (g > 0.0f) {
            int pos = atomicAdd(&counts[e], 1);
            list[e * NTOK + pos]    = n;
            wlist[e * NTOK + pos]   = g / denom[e] * CAPACITY;
            slotArr[e * NTOK + pos] = slot++;
        }
    }
}

// ---------------------------------------------------------------------------
// K4: fused expert FFN.  Workgroup = 64 tokens of one expert, 512 threads
// (16 waves), ~199KB LDS.  Per chunk of 512 INNER columns:
//   GEMM1: wave owns 2 col-tiles; B loaded once per k-step, reused across
//          the 4 row-tiles; bias + erf-GELU into LDS bf16.
//   GEMM2: wave owns 4 col-tiles x 4 row-tiles (acc[16]); A fragments built
//          once per k-step and reused across the 4 B tiles.
// WT = __bf16: weights are pre-swizzled WMMA-B tiles -> one 32B vector load
//              per lane per tile (2x global_load_b128, fully coalesced).
// WT = float:  row-major streaming fallback (scalar gather + cvt).
// ---------------------------------------------------------------------------
template <typename WT>
__launch_bounds__(THREADS, 1)
__global__ void ffn_kernel(const WT* __restrict__ x,
                           const WT* __restrict__ W1,
                           const float* __restrict__ b1,
                           const WT* __restrict__ W2,
                           const float* __restrict__ b2,
                           const int* __restrict__ counts,
                           const int* __restrict__ list,
                           const float* __restrict__ wlist,
                           const int* __restrict__ slotArr,
                           float* __restrict__ buf /* [2][NTOK][OUTD] */) {
    constexpr bool BF = std::is_same<WT, __bf16>::value;

    __shared__ __bf16 xs[TM][DIM + XPAD];   // 129.0 KB
    __shared__ __bf16 hs[TM][CI + HPAD];    //  65.0 KB
    __shared__ int   s_tok[TM];
    __shared__ float s_gw[TM];
    __shared__ int   s_slot[TM];
    __shared__ int   s_val[TM];

    int e   = blockIdx.x >> 6;        // 64 token-blocks per expert
    int blk = blockIdx.x & 63;
    int cnt = counts[e];
    int r0  = blk * TM;
    if (r0 >= cnt) return;            // block-uniform: safe w.r.t. WMMA EXEC

    int tid   = threadIdx.x;
    int wave  = tid >> 5;             // 0..15
    int lane  = tid & 31;
    int halfw = lane >> 4;
    int nn    = lane & 15;

    if (tid < TM) {
        int r = r0 + tid;
        int valid = (r < cnt) ? 1 : 0;
        s_tok[tid]  = valid ? list[e * NTOK + r] : list[e * NTOK]; // cnt>=1
        s_gw[tid]   = valid ? wlist[e * NTOK + r] : 0.0f;
        s_slot[tid] = valid ? slotArr[e * NTOK + r] : 0;
        s_val[tid]  = valid;
    }
    __syncthreads();

    // stage x tile into bf16 LDS
    if constexpr (BF) {
        for (int i = tid; i < TM * DIM / 2; i += THREADS) {
            int m = i >> 9;
            int d = i & (DIM / 2 - 1);
            ((unsigned*)&xs[m][0])[d] =
                ((const unsigned*)(x + (size_t)s_tok[m] * DIM))[d];
        }
    } else {
        for (int i = tid; i < TM * DIM; i += THREADS) {
            int m = i >> 10;
            int d = i & (DIM - 1);
            xs[m][d] = (__bf16)x[(size_t)s_tok[m] * DIM + d];
        }
    }
    __syncthreads();

    // weight bases: tiled (bf16) or row-major (float)
    const WT* W1e = W1 + (size_t)e * DIM * INNER;   // same elem count either way
    const WT* W2e = W2 + (size_t)e * INNER * OUTD;

    v8f acc[16];
#pragma unroll
    for (int t = 0; t < 16; ++t)
        acc[t] = (v8f){0.f, 0.f, 0.f, 0.f, 0.f, 0.f, 0.f, 0.f};

    for (int c = 0; c < INNER / CI; ++c) {          // 8 chunks of 512
        int cBase = c * CI;

        // ---- GEMM1: h[64, CI] = gelu(xs @ W1[:, chunk] + b1) ----
        for (int cc = 0; cc < 2; ++cc) {            // 32 col-tiles / 16 waves
            int lcol = (wave * 2 + cc) * 16;
            int gcol = cBase + lcol;
            v8f hacc[4];
#pragma unroll
            for (int rt = 0; rt < 4; ++rt)
                hacc[rt] = (v8f){0.f, 0.f, 0.f, 0.f, 0.f, 0.f, 0.f, 0.f};
            for (int k0 = 0; k0 < DIM; k0 += 32) {
                v16bf b;
                if constexpr (BF) {
                    size_t tile = ((size_t)(k0 >> 5) * (INNER >> 4) + (gcol >> 4));
                    b = *(const v16bf*)(W1e + (tile << 9) + (lane << 4));
                } else {
                    b = glob_frag_b<WT>(W1e + gcol + nn, k0, INNER, halfw);
                }
#pragma unroll
                for (int rt = 0; rt < 4; ++rt) {    // reuse B across row-tiles
                    v16bf a = lds_frag_a(&xs[(rt << 4) + nn][0], k0, halfw);
                    hacc[rt] = __builtin_amdgcn_wmma_f32_16x16x32_bf16(
                        false, a, false, b, (short)0, hacc[rt], false, false);
                }
            }
            float bia = b1[e * INNER + gcol + nn];
#pragma unroll
            for (int rt = 0; rt < 4; ++rt)
#pragma unroll
                for (int r = 0; r < 8; ++r) {
                    int m = (rt << 4) + r + (halfw << 3);   // C/D: M=r+8*half
                    hs[m][lcol + nn] = (__bf16)gelu_erf(hacc[rt][r] + bia);
                }
        }
        __syncthreads();

        // ---- GEMM2: acc += hs[64, CI] @ W2[chunk, :OUTD] ----
        for (int k0 = 0; k0 < CI; k0 += 32) {
            v16bf a[4];
#pragma unroll
            for (int rt = 0; rt < 4; ++rt)          // A fragments built once
                a[rt] = lds_frag_a(&hs[(rt << 4) + nn][0], k0, halfw);
#pragma unroll
            for (int cc = 0; cc < 4; ++cc) {        // 64 col-tiles / 16 waves
                int ocol = (wave * 4 + cc) * 16;
                v16bf b;
                if constexpr (BF) {
                    size_t tile = ((size_t)((cBase + k0) >> 5) * (OUTD >> 4) +
                                   (ocol >> 4));
                    b = *(const v16bf*)(W2e + (tile << 9) + (lane << 4));
                } else {
                    b = glob_frag_b<WT>(W2e + ocol + nn, cBase + k0, OUTD, halfw);
                }
#pragma unroll
                for (int rt = 0; rt < 4; ++rt)      // reuse B across row-tiles
                    acc[cc * 4 + rt] = __builtin_amdgcn_wmma_f32_16x16x32_bf16(
                        false, a[rt], false, b, (short)0, acc[cc * 4 + rt],
                        false, false);
            }
        }
        __syncthreads();    // before next chunk overwrites hs
    }

    // ---- epilogue: buf[slot][tok][col] = gw * (acc + b2) ----
    for (int cc = 0; cc < 4; ++cc) {
        int ocol = (wave * 4 + cc) * 16;
        float bb = b2[e * OUTD + ocol + nn];
#pragma unroll
        for (int rt = 0; rt < 4; ++rt)
#pragma unroll
            for (int r = 0; r < 8; ++r) {
                int m = (rt << 4) + r + (halfw << 3);
                if (s_val[m]) {
                    float vv = s_gw[m] * (acc[cc * 4 + rt][r] + bb);
                    buf[((size_t)s_slot[m] * NTOK + s_tok[m]) * OUTD + ocol + nn] = vv;
                }
            }
    }
}

// ---------------------------------------------------------------------------
// K5: combine the two per-slot contributions (deterministic, no atomics).
// ---------------------------------------------------------------------------
__global__ void combine_kernel(const float* __restrict__ buf,
                               float* __restrict__ out) {
    long long total = (long long)NTOK * OUTD;
    long long i = (long long)blockIdx.x * 256 + threadIdx.x;
    if (i < total) out[i] = buf[i] + buf[total + i];
}

// ---------------------------------------------------------------------------
extern "C" void kernel_launch(void* const* d_in, const int* in_sizes, int n_in,
                              void* d_out, int out_size, void* d_ws, size_t ws_size,
                              hipStream_t stream) {
    const float* x  = (const float*)d_in[0];
    const float* gi = (const float*)d_in[1];
    const float* W1 = (const float*)d_in[2];
    const float* b1 = (const float*)d_in[3];
    const float* W2 = (const float*)d_in[4];
    const float* b2 = (const float*)d_in[5];
    const float* Wg = (const float*)d_in[6];
    const float* bg = (const float*)d_in[7];
    float* out = (float*)d_out;

    const long long W_ELEMS = (long long)NEXP * DIM * INNER;   // = E*INNER*OUTD
    const long long X_ELEMS = (long long)NTOK * DIM;
    const long long BUF_ELEMS = 2LL * NTOK * OUTD;

    // workspace layout
    char* ws = (char*)d_ws;
    float* gmask   = (float*)ws;  ws += (size_t)NTOK * NEXP * sizeof(float);
    float* denom   = (float*)ws;  ws += 64;
    int*   counts  = (int*)ws;    ws += 64;
    int*   list    = (int*)ws;    ws += (size_t)NEXP * NTOK * sizeof(int);
    float* wlist   = (float*)ws;  ws += (size_t)NEXP * NTOK * sizeof(float);
    int*   slotArr = (int*)ws;    ws += (size_t)NEXP * NTOK * sizeof(int);
    float* buf     = (float*)ws;  ws += (size_t)BUF_ELEMS * sizeof(float);
    __bf16* xbf    = (__bf16*)ws; ws += (size_t)X_ELEMS * sizeof(__bf16);
    __bf16* w1t    = (__bf16*)ws; ws += (size_t)W_ELEMS * sizeof(__bf16);
    __bf16* w2t    = (__bf16*)ws; ws += (size_t)W_ELEMS * sizeof(__bf16);
    size_t need_full = (size_t)(ws - (char*)d_ws);
    bool preconv = (ws_size >= need_full);

    init_kernel<<<4096, 256, 0, stream>>>(counts, buf, BUF_ELEMS);
    gate_kernel<<<NTOK / 8, 256, 0, stream>>>(gi, Wg, bg, gmask);
    denom_kernel<<<NEXP, 256, 0, stream>>>(gmask, denom);
    route_kernel<<<NTOK / 256, 256, 0, stream>>>(gmask, denom, counts, list,
                                                 wlist, slotArr);
    if (preconv) {
        cvt_tile_kernel<<<NEXP * (DIM / 32), 256, 0, stream>>>(W1, w1t,
                                                               DIM, INNER);
        cvt_tile_kernel<<<NEXP * (INNER / 32), 256, 0, stream>>>(W2, w2t,
                                                                 INNER, OUTD);
        cvt_bf16_kernel<<<2048, 256, 0, stream>>>(x, xbf, X_ELEMS);
        ffn_kernel<__bf16><<<NEXP * (NTOK / TM), THREADS, 0, stream>>>(
            xbf, w1t, b1, w2t, b2, counts, list, wlist, slotArr, buf);
    } else {
        ffn_kernel<float><<<NEXP * (NTOK / TM), THREADS, 0, stream>>>(
            x, W1, b1, W2, b2, counts, list, wlist, slotArr, buf);
    }
    combine_kernel<<<(int)((BUF_ELEMS / 2 + 255) / 256), 256, 0,
                     stream>>>(buf, out);
}
